// RelationalSelfAttention_19490561589529
// MI455X (gfx1250) — matrix operand order repaired and checked by reading, not compiled, not run on hardware
//
#include <hip/hip_runtime.h>

// Problem constants
constexpr int BB   = 2;
constexpr int NN   = 2048;
constexpr int DIMM = 1024;
constexpr int HH   = 16;
constexpr int EE   = 8;
constexpr int HD   = 64;
constexpr float SCALE = 0.125f; // HD^-0.5

typedef __attribute__((ext_vector_type(16))) _Float16 v16h;
typedef __attribute__((ext_vector_type(8)))  _Float16 v8h;
typedef __attribute__((ext_vector_type(8)))  float    v8f;

__device__ __forceinline__ v8f wmma_f16(v16h a, v16h b, v8f c) {
  // D = A(16x32 f16) * B(32x16 f16) + C(16x16 f32)
  return __builtin_amdgcn_wmma_f32_16x16x32_f16(
      /*neg_a=*/false, a, /*neg_b=*/false, b,
      /*c_mod=*/(short)0, c, /*reuse_a=*/false, /*reuse_b=*/false);
}

// A-fragment: 16x32 f16, row-major source, lane=row (mod 16).
// halves 0..15 = K k0+0..15 (lanes 0-15) / k0+16..31 (lanes 16-31)
__device__ __forceinline__ v16h load_a_frag(const _Float16* __restrict__ base,
                                            int stride, int row0, int k0) {
  const int lane = threadIdx.x & 31;
  const int li = lane & 15, hi4 = lane >> 4;
  const _Float16* p = base + (size_t)(row0 + li) * stride + k0 + hi4 * 16;
  union { v16h v; v8h h[2]; } u;
  u.h[0] = *(const v8h*)(p);
  u.h[1] = *(const v8h*)(p + 8);
  return u.v;
}

// B-fragment: 32x16 f16, columns are rows of a row-major [Ncols, K] matrix.
// lane = column (mod 16); 16 contiguous K halves: k0+0..15 / k0+16..31.
__device__ __forceinline__ v16h load_b_frag(const _Float16* __restrict__ base,
                                            int stride, int col0, int k0) {
  const int lane = threadIdx.x & 31;
  const int li = lane & 15, hi4 = lane >> 4;
  const _Float16* p = base + (size_t)(col0 + li) * stride + k0 + hi4 * 16;
  return *(const v16h*)p;
}

__global__ void cast_f32_to_f16(const float* __restrict__ src,
                                _Float16* __restrict__ dst, int n) {
  int i = blockIdx.x * blockDim.x + threadIdx.x;
  if (i < n) dst[i] = (_Float16)src[i];
}

// ---------------------------------------------------------------------------
// Kernel 1: QKV GEMM.  out[n,d] = sum_c x[n,c]*w_qkv[d,c] + b_qkv[d]
// Each wave: 32 rows x 64 cols (B-fragments reused across both row halves).
// Scatter into q[B,H,N,HD]*SCALE, k[B,H,N,HD], vT[B,H,HD,N].
// ---------------------------------------------------------------------------
__global__ __launch_bounds__(256) void qkv_gemm(
    const _Float16* __restrict__ xh, const _Float16* __restrict__ wh,
    const float* __restrict__ bq, _Float16* __restrict__ qh,
    _Float16* __restrict__ kh, _Float16* __restrict__ vt) {
  const int wave = threadIdx.x >> 5;
  const int gw = blockIdx.x * 8 + wave;          // 6144 wave-tiles
  const int mtile = gw / 48;                     // 0..127  (32 rows each)
  const int nt = gw - mtile * 48;                // 0..47   (64 cols each)
  const int row0 = mtile * 32;
  const int col0 = nt * 64;
  const int lane = threadIdx.x & 31;
  const int li = lane & 15, hi4 = lane >> 4;

  const v8f vzero = {0.f,0.f,0.f,0.f,0.f,0.f,0.f,0.f};
  v8f accA[4], accB[4];
#pragma unroll
  for (int t = 0; t < 4; ++t) { accA[t] = vzero; accB[t] = vzero; }

  for (int k = 0; k < DIMM; k += 32) {
    if (k + 64 < DIMM)
      __builtin_prefetch((const void*)(xh + (size_t)(row0 + li) * DIMM + k + 64), 0, 1);
    v16h a0 = load_a_frag(xh, DIMM, row0,      k);
    v16h a1 = load_a_frag(xh, DIMM, row0 + 16, k);
#pragma unroll
    for (int t = 0; t < 4; ++t) {
      v16h b = load_b_frag(wh, DIMM, col0 + t * 16, k);
      accA[t] = wmma_f16(a0, b, accA[t]);
      accB[t] = wmma_f16(a1, b, accB[t]);
    }
  }

  const int which = col0 >> 10;        // 0=q 1=k 2=v (64-col tile never straddles)
  const int h = (col0 & 1023) >> 6;    // head
  const int bidx = row0 >> 11;         // batch (32-row tile never straddles)
  const int bh = bidx * HH + h;
#pragma unroll
  for (int t = 0; t < 4; ++t) {
    const int hd = t * 16 + li;
    const float bv = bq[col0 + t * 16 + li];
#pragma unroll
    for (int r = 0; r < 8; ++r) {
      const int n0 = (row0 & (NN - 1)) + r + hi4 * 8;
      const float vA = accA[t][r] + bv;
      const float vB = accB[t][r] + bv;
      if (which == 0) {
        qh[((size_t)bh * NN + n0) * HD + hd]      = (_Float16)(vA * SCALE);
        qh[((size_t)bh * NN + n0 + 16) * HD + hd] = (_Float16)(vB * SCALE);
      } else if (which == 1) {
        kh[((size_t)bh * NN + n0) * HD + hd]      = (_Float16)vA;
        kh[((size_t)bh * NN + n0 + 16) * HD + hd] = (_Float16)vB;
      } else {
        vt[((size_t)bh * HD + hd) * NN + n0]      = (_Float16)vA;
        vt[((size_t)bh * HD + hd) * NN + n0 + 16] = (_Float16)vB;
      }
    }
  }
}

// ---------------------------------------------------------------------------
// Kernel 2: fused flash attention per (b,h,16-row tile), 64 keys per step.
// S = q k^T (scale pre-folded) + rel_bias, online softmax, O += P v.
// rel_bias[h, i%8, j%8]: all sub-tiles 16-aligned -> bias = f(r, lane&7) only.
// ---------------------------------------------------------------------------
__global__ __launch_bounds__(128) void flash_attn(
    const _Float16* __restrict__ qh, const _Float16* __restrict__ kh,
    const _Float16* __restrict__ vt, const float* __restrict__ rel_pos,
    _Float16* __restrict__ oh) {
  __shared__ __align__(32) _Float16 lds_p[4][16][64];

  const int wave = threadIdx.x >> 5;
  const int gw = blockIdx.x * 4 + wave;  // 4096 waves total
  const int rt = gw & 127;               // row tile within (b,h)
  const int bh = gw >> 7;                // 0..31
  const int bidx = bh / HH, h = bh % HH;
  const int row0 = rt * 16;
  const int lane = threadIdx.x & 31;
  const int li = lane & 15, hi4 = lane >> 4;

  const _Float16* qb = qh + (size_t)bh * NN * HD;
  const _Float16* kb = kh + (size_t)bh * NN * HD;
  const _Float16* vb = vt + (size_t)bh * HD * NN;

  const v16h qf0 = load_a_frag(qb, HD, row0, 0);
  const v16h qf1 = load_a_frag(qb, HD, row0, 32);

  float bias[8];
#pragma unroll
  for (int r = 0; r < 8; ++r)
    bias[r] = rel_pos[(h * EE + r) * EE + (lane & 7)];

  const v8f vzero = {0.f,0.f,0.f,0.f,0.f,0.f,0.f,0.f};
  v8f O[4];
#pragma unroll
  for (int t = 0; t < 4; ++t) O[t] = vzero;
  float m[8], l[8];
#pragma unroll
  for (int r = 0; r < 8; ++r) { m[r] = -1e30f; l[r] = 0.f; }

  for (int nb = 0; nb < NN; nb += 64) {
    // scores: 16 rows x 64 keys
    v8f S[4];
#pragma unroll
    for (int c = 0; c < 4; ++c) {
      S[c] = vzero;
      S[c] = wmma_f16(qf0, load_b_frag(kb, HD, nb + c * 16, 0),  S[c]);
      S[c] = wmma_f16(qf1, load_b_frag(kb, HD, nb + c * 16, 32), S[c]);
    }

#pragma unroll
    for (int r = 0; r < 8; ++r) {
      float s0 = S[0][r] + bias[r];
      float s1 = S[1][r] + bias[r];
      float s2 = S[2][r] + bias[r];
      float s3 = S[3][r] + bias[r];
      // row max across the 16 lanes that share this row (wave32 halves
      // reduce independently for xor masks < 16)
      float mt = fmaxf(fmaxf(s0, s1), fmaxf(s2, s3));
      mt = fmaxf(mt, __shfl_xor(mt, 1));
      mt = fmaxf(mt, __shfl_xor(mt, 2));
      mt = fmaxf(mt, __shfl_xor(mt, 4));
      mt = fmaxf(mt, __shfl_xor(mt, 8));
      float mnew = fmaxf(m[r], mt);
      float resc = __expf(m[r] - mnew);
      float p0 = __expf(s0 - mnew);
      float p1 = __expf(s1 - mnew);
      float p2 = __expf(s2 - mnew);
      float p3 = __expf(s3 - mnew);
      float rs = (p0 + p1) + (p2 + p3);
      rs += __shfl_xor(rs, 1);
      rs += __shfl_xor(rs, 2);
      rs += __shfl_xor(rs, 4);
      rs += __shfl_xor(rs, 8);
      l[r] = l[r] * resc + rs;
      m[r] = mnew;
      O[0][r] *= resc; O[1][r] *= resc; O[2][r] *= resc; O[3][r] *= resc;
      // stage P (C-fragment layout -> row-major LDS) for A-fragment reload
      const int pr = r + hi4 * 8;
      lds_p[wave][pr][li]      = (_Float16)p0;
      lds_p[wave][pr][li + 16] = (_Float16)p1;
      lds_p[wave][pr][li + 32] = (_Float16)p2;
      lds_p[wave][pr][li + 48] = (_Float16)p3;
    }
    asm volatile("s_wait_dscnt 0" ::: "memory");
    v16h pf0, pf1;
    {
      union { v16h v; v8h h2[2]; } u0, u1;
      const _Float16* lp0 = &lds_p[wave][li][hi4 * 16];
      const _Float16* lp1 = &lds_p[wave][li][32 + hi4 * 16];
      u0.h2[0] = *(const v8h*)(lp0);
      u0.h2[1] = *(const v8h*)(lp0 + 8);
      u1.h2[0] = *(const v8h*)(lp1);
      u1.h2[1] = *(const v8h*)(lp1 + 8);
      pf0 = u0.v; pf1 = u1.v;
    }
    // O(16x64) += P(16x64) * V(64x64); B-fragments contiguous from vT
#pragma unroll
    for (int t = 0; t < 4; ++t) {
      O[t] = wmma_f16(pf0, load_b_frag(vb, NN, t * 16, nb),      O[t]);
      O[t] = wmma_f16(pf1, load_b_frag(vb, NN, t * 16, nb + 32), O[t]);
    }
  }

#pragma unroll
  for (int r = 0; r < 8; ++r) {
    const float inv = 1.f / l[r];
    const int n = row0 + r + hi4 * 8;
    const size_t base = ((size_t)bidx * NN + n) * DIMM + h * HD;
    oh[base + 0  + li] = (_Float16)(O[0][r] * inv);
    oh[base + 16 + li] = (_Float16)(O[1][r] * inv);
    oh[base + 32 + li] = (_Float16)(O[2][r] * inv);
    oh[base + 48 + li] = (_Float16)(O[3][r] * inv);
  }
}

// ---------------------------------------------------------------------------
// Kernel 3: projection GEMM. out[n,d] = sum_c a[n,c]*w_proj[d,c] + b_proj[d]
// Each wave: 32 rows x 64 cols.
// ---------------------------------------------------------------------------
__global__ __launch_bounds__(256) void proj_gemm(
    const _Float16* __restrict__ ah, const _Float16* __restrict__ wh,
    const float* __restrict__ bp, float* __restrict__ out) {
  const int wave = threadIdx.x >> 5;
  const int gw = blockIdx.x * 8 + wave;   // 2048 wave-tiles
  const int mtile = gw >> 4;              // 0..127 (32 rows)
  const int nt = gw & 15;                 // 64 cols
  const int row0 = mtile * 32;
  const int col0 = nt * 64;
  const int lane = threadIdx.x & 31;
  const int li = lane & 15, hi4 = lane >> 4;

  const v8f vzero = {0.f,0.f,0.f,0.f,0.f,0.f,0.f,0.f};
  v8f accA[4], accB[4];
#pragma unroll
  for (int t = 0; t < 4; ++t) { accA[t] = vzero; accB[t] = vzero; }

  for (int k = 0; k < DIMM; k += 32) {
    if (k + 64 < DIMM)
      __builtin_prefetch((const void*)(ah + (size_t)(row0 + li) * DIMM + k + 64), 0, 1);
    v16h a0 = load_a_frag(ah, DIMM, row0,      k);
    v16h a1 = load_a_frag(ah, DIMM, row0 + 16, k);
#pragma unroll
    for (int t = 0; t < 4; ++t) {
      v16h b = load_b_frag(wh, DIMM, col0 + t * 16, k);
      accA[t] = wmma_f16(a0, b, accA[t]);
      accB[t] = wmma_f16(a1, b, accB[t]);
    }
  }

#pragma unroll
  for (int t = 0; t < 4; ++t) {
    const int colg = col0 + t * 16 + li;
    const float bv = bp[colg];
#pragma unroll
    for (int r = 0; r < 8; ++r) {
      const int row = row0 + r + hi4 * 8;
      out[(size_t)row * DIMM + colg]        = accA[t][r] + bv;
      out[(size_t)(row + 16) * DIMM + colg] = accB[t][r] + bv;
    }
  }
}

// ---------------------------------------------------------------------------
extern "C" void kernel_launch(void* const* d_in, const int* in_sizes, int n_in,
                              void* d_out, int out_size, void* d_ws, size_t ws_size,
                              hipStream_t stream) {
  const float* x      = (const float*)d_in[0];
  const float* w_qkv  = (const float*)d_in[1];
  const float* b_qkv  = (const float*)d_in[2];
  const float* w_proj = (const float*)d_in[3];
  const float* b_proj = (const float*)d_in[4];
  const float* rel    = (const float*)d_in[5];

  char* ws = (char*)d_ws;
  _Float16* x_h     = (_Float16*)(ws + 0);         //  8 MB: [B*N, DIM]
  _Float16* wqkv_h  = (_Float16*)(ws + 8388608);   //  6 MB: [3*DIM, DIM]
  _Float16* wproj_h = (_Float16*)(ws + 14680064);  //  2 MB: [DIM, DIM]
  _Float16* q_h     = (_Float16*)(ws + 16777216);  //  8 MB: [B,H,N,HD] (pre-scaled)
  _Float16* k_h     = (_Float16*)(ws + 25165824);  //  8 MB: [B,H,N,HD]
  _Float16* v_t     = (_Float16*)(ws + 33554432);  //  8 MB: [B,H,HD,N]
  _Float16* attn_h  = (_Float16*)(ws + 41943040);  //  8 MB: [B*N, DIM]

  const int n_x  = BB * NN * DIMM;        // 4194304
  const int n_wq = 3 * DIMM * DIMM;       // 3145728
  const int n_wp = DIMM * DIMM;           // 1048576

  cast_f32_to_f16<<<(n_x  + 255) / 256, 256, 0, stream>>>(x,      x_h,     n_x);
  cast_f32_to_f16<<<(n_wq + 255) / 256, 256, 0, stream>>>(w_qkv,  wqkv_h,  n_wq);
  cast_f32_to_f16<<<(n_wp + 255) / 256, 256, 0, stream>>>(w_proj, wproj_h, n_wp);

  // 128 m-tiles(32 rows) * 48 n-tiles(64 cols) = 6144 waves / 8 per block
  qkv_gemm<<<768, 256, 0, stream>>>(x_h, wqkv_h, b_qkv, q_h, k_h, v_t);

  // B*H*(N/16) = 4096 waves / 4 per block
  flash_attn<<<1024, 128, 0, stream>>>(q_h, k_h, v_t, rel, attn_h);

  // 128 m-tiles * 16 n-tiles = 2048 waves / 8 per block
  proj_gemm<<<256, 256, 0, stream>>>(attn_h, wproj_h, b_proj, (float*)d_out);
}
